// TRANSFORMERLAYER_4758823764049
// MI455X (gfx1250) — matrix operand order repaired and checked by reading, not compiled, not run on hardware
//
#include <hip/hip_runtime.h>

typedef __attribute__((ext_vector_type(16))) __bf16 v16bf;
typedef __attribute__((ext_vector_type(8)))  __bf16 v8bf;
typedef __attribute__((ext_vector_type(4)))  __bf16 v4bf;
typedef __attribute__((ext_vector_type(8)))  float  v8f;
typedef __attribute__((ext_vector_type(4)))  float  f32x4;

// ---------- fragment builders ----------

// From f32 memory: two 8-float contiguous chunks -> 16 bf16 (fallback path).
__device__ __forceinline__ v16bf frag_f32_2x8(const float* __restrict__ p0,
                                              const float* __restrict__ p1) {
  f32x4 a0 = *(const f32x4*)(p0);
  f32x4 a1 = *(const f32x4*)(p0 + 4);
  f32x4 a2 = *(const f32x4*)(p1);
  f32x4 a3 = *(const f32x4*)(p1 + 4);
  v16bf r;
  r[0]  = (__bf16)a0[0]; r[1]  = (__bf16)a0[1]; r[2]  = (__bf16)a0[2]; r[3]  = (__bf16)a0[3];
  r[4]  = (__bf16)a1[0]; r[5]  = (__bf16)a1[1]; r[6]  = (__bf16)a1[2]; r[7]  = (__bf16)a1[3];
  r[8]  = (__bf16)a2[0]; r[9]  = (__bf16)a2[1]; r[10] = (__bf16)a2[2]; r[11] = (__bf16)a2[3];
  r[12] = (__bf16)a3[0]; r[13] = (__bf16)a3[1]; r[14] = (__bf16)a3[2]; r[15] = (__bf16)a3[3];
  return r;
}

// From bf16 memory: two 8-bf16 contiguous chunks (16B each) -> 16 bf16.
__device__ __forceinline__ v16bf frag_bf16_2x8(const __bf16* __restrict__ p0,
                                               const __bf16* __restrict__ p1) {
  v8bf a = *(const v8bf*)(p0);
  v8bf b = *(const v8bf*)(p1);
  v16bf r;
#pragma unroll
  for (int i = 0; i < 8; ++i) { r[i] = a[i]; r[i + 8] = b[i]; }
  return r;
}

__device__ __forceinline__ float gelu_tanh(float v) {
  // jax.nn.gelu default: approximate=True (tanh form)
  float t = tanhf(0.7978845608028654f * (v + 0.044715f * v * v * v));
  return 0.5f * v * (1.0f + t);
}

// ---------- f32 -> bf16 pre-conversion ----------
__global__ __launch_bounds__(256)
void NEURONMEAN_cvt_kernel(const float* __restrict__ src,
                           __bf16* __restrict__ dst, int count) {
  int i = (blockIdx.x * blockDim.x + threadIdx.x) * 4;
  if (i >= count) return;                    // count is a multiple of 4
  f32x4 v = *(const f32x4*)(src + i);
  v4bf o;
  o[0] = (__bf16)v[0]; o[1] = (__bf16)v[1];
  o[2] = (__bf16)v[2]; o[3] = (__bf16)v[3];
  *(v4bf*)(dst + i) = o;
}

// ---------- GEMM + fused gelu/column-sum ----------
// Block = 256 threads = 8 waves, 4 (M-dir) x 2 (N-dir).
// Block tile: 256 rows (T) x 64 cols (N). Wave tile: 64 x 32 (eight 16x16 accs).
// partials[slot][n], slot = blockIdx.y*4 + waveM  (T/64 slots total).
// Epilogue shared by both variants.
__device__ __forceinline__ void epilogue(v8f acc[4][2], const float* __restrict__ b,
                                         float* __restrict__ partials,
                                         int nBase, int l16, int lane, int waveM, int N) {
#pragma unroll
  for (int n2 = 0; n2 < 2; ++n2) {
    const int n = nBase + n2 * 16 + l16;
    const float bb = b[n];
    float colsum = 0.0f;
#pragma unroll
    for (int m = 0; m < 4; ++m) {
#pragma unroll
      for (int r = 0; r < 8; ++r) {
        colsum += gelu_tanh(acc[m][n2][r] + bb);
      }
    }
    // combine M=0..7 (lanes 0-15) with M=8..15 (lanes 16-31) for same n
    colsum += __shfl_xor(colsum, 16, 32);
    if (lane < 16) {
      const int slot = blockIdx.y * 4 + waveM;
      partials[(size_t)slot * (size_t)N + n] = colsum;
    }
  }
}

// Fast path: inputs pre-converted to bf16 in workspace.
__global__ __launch_bounds__(256)
void NEURONMEAN_gemm_bf16_kernel(const __bf16* __restrict__ x,
                                 const __bf16* __restrict__ W,
                                 const float* __restrict__ b,
                                 float* __restrict__ partials,
                                 int D, int N) {
  const int lane  = threadIdx.x & 31;
  const int wid   = threadIdx.x >> 5;
  const int waveM = wid & 3;
  const int waveN = wid >> 2;
  const int half  = lane >> 4;
  const int l16   = lane & 15;

  const int tBase = blockIdx.y * 256 + waveM * 64;
  const int nBase = blockIdx.x * 64  + waveN * 32;

  v8f acc[4][2] = {};

  const __bf16* xrow[4];
#pragma unroll
  for (int m = 0; m < 4; ++m)
    xrow[m] = x + (size_t)(tBase + m * 16 + l16) * (size_t)D;
  const __bf16* wrow[2];
#pragma unroll
  for (int n2 = 0; n2 < 2; ++n2)
    wrow[n2] = W + (size_t)(nBase + n2 * 16 + l16) * (size_t)D;

  const int aoff = 8 * half;    // A frag: chunks at k+aoff and k+16+aoff
  const int boff = 8 * half;    // B frag: chunks at k+boff and k+16+boff

  for (int k = 0; k < D; k += 32) {
    v16bf bf0 = frag_bf16_2x8(wrow[0] + k + boff, wrow[0] + k + 16 + boff);
    v16bf bf1 = frag_bf16_2x8(wrow[1] + k + boff, wrow[1] + k + 16 + boff);
#pragma unroll
    for (int m = 0; m < 4; ++m) {
      v16bf af = frag_bf16_2x8(xrow[m] + k + aoff, xrow[m] + k + 16 + aoff);
      acc[m][0] = __builtin_amdgcn_wmma_f32_16x16x32_bf16(
          false, af, false, bf0, (short)0, acc[m][0], false, false);
      acc[m][1] = __builtin_amdgcn_wmma_f32_16x16x32_bf16(
          false, af, false, bf1, (short)0, acc[m][1], false, false);
    }
  }

  epilogue(acc, b, partials, nBase, l16, lane, waveM, N);
}

// Fallback path: f32 inputs, convert in-register (used if workspace too small).
__global__ __launch_bounds__(256)
void NEURONMEAN_gemm_f32_kernel(const float* __restrict__ x,
                                const float* __restrict__ W,
                                const float* __restrict__ b,
                                float* __restrict__ partials,
                                int D, int N) {
  const int lane  = threadIdx.x & 31;
  const int wid   = threadIdx.x >> 5;
  const int waveM = wid & 3;
  const int waveN = wid >> 2;
  const int half  = lane >> 4;
  const int l16   = lane & 15;

  const int tBase = blockIdx.y * 256 + waveM * 64;
  const int nBase = blockIdx.x * 64  + waveN * 32;

  v8f acc[4][2] = {};

  const float* xrow[4];
#pragma unroll
  for (int m = 0; m < 4; ++m)
    xrow[m] = x + (size_t)(tBase + m * 16 + l16) * (size_t)D;
  const float* wrow[2];
#pragma unroll
  for (int n2 = 0; n2 < 2; ++n2)
    wrow[n2] = W + (size_t)(nBase + n2 * 16 + l16) * (size_t)D;

  const int aoff = 8 * half;
  const int boff = 8 * half;

  for (int k = 0; k < D; k += 32) {
    v16bf bf0 = frag_f32_2x8(wrow[0] + k + boff, wrow[0] + k + 16 + boff);
    v16bf bf1 = frag_f32_2x8(wrow[1] + k + boff, wrow[1] + k + 16 + boff);
#pragma unroll
    for (int m = 0; m < 4; ++m) {
      v16bf af = frag_f32_2x8(xrow[m] + k + aoff, xrow[m] + k + 16 + aoff);
      acc[m][0] = __builtin_amdgcn_wmma_f32_16x16x32_bf16(
          false, af, false, bf0, (short)0, acc[m][0], false, false);
      acc[m][1] = __builtin_amdgcn_wmma_f32_16x16x32_bf16(
          false, af, false, bf1, (short)0, acc[m][1], false, false);
    }
  }

  epilogue(acc, b, partials, nBase, l16, lane, waveM, N);
}

// ---------- final reduction ----------
__global__ __launch_bounds__(256)
void NEURONMEAN_reduce_kernel(const float* __restrict__ partials,
                              float* __restrict__ out,
                              int N, int S, float invT) {
  const int n = blockIdx.x * blockDim.x + threadIdx.x;
  if (n >= N) return;
  float s = 0.0f;
  for (int i = 0; i < S; ++i) s += partials[(size_t)i * (size_t)N + n];
  out[n] = s * invT;
}

extern "C" void kernel_launch(void* const* d_in, const int* in_sizes, int n_in,
                              void* d_out, int out_size, void* d_ws, size_t ws_size,
                              hipStream_t stream) {
  const float* x = (const float*)d_in[0];  // [T, D]
  const float* W = (const float*)d_in[1];  // [N, D]
  const float* b = (const float*)d_in[2];  // [N]
  float* out = (float*)d_out;              // [1, N]

  const int N = in_sizes[2];
  const int D = in_sizes[1] / N;
  const int T = in_sizes[0] / D;
  const int S = (T / 256) * 4;             // partial slots = T/64

  const size_t xElems = (size_t)T * (size_t)D;
  const size_t wElems = (size_t)N * (size_t)D;
  const size_t bf16Bytes = (xElems + wElems) * sizeof(__bf16);
  const size_t partBytes = (size_t)S * (size_t)N * sizeof(float);

  dim3 grid(N / 64, T / 256);

  if (ws_size >= bf16Bytes + partBytes) {
    // Fast path: pre-convert x and W to bf16 once per launch, then pure-bf16 GEMM.
    __bf16* xbf = (__bf16*)d_ws;
    __bf16* wbf = xbf + xElems;
    float* partials = (float*)((char*)d_ws + bf16Bytes);

    NEURONMEAN_cvt_kernel<<<(int)((xElems / 4 + 255) / 256), 256, 0, stream>>>(
        x, xbf, (int)xElems);
    NEURONMEAN_cvt_kernel<<<(int)((wElems / 4 + 255) / 256), 256, 0, stream>>>(
        W, wbf, (int)wElems);
    NEURONMEAN_gemm_bf16_kernel<<<grid, 256, 0, stream>>>(
        xbf, wbf, b, partials, D, N);
    NEURONMEAN_reduce_kernel<<<(N + 255) / 256, 256, 0, stream>>>(
        partials, out, N, S, 1.0f / (float)T);
  } else {
    // Fallback: convert in-register inside the GEMM.
    float* partials = (float*)d_ws;
    NEURONMEAN_gemm_f32_kernel<<<grid, 256, 0, stream>>>(
        x, W, b, partials, D, N);
    NEURONMEAN_reduce_kernel<<<(N + 255) / 256, 256, 0, stream>>>(
        partials, out, N, S, 1.0f / (float)T);
  }
}
